// ProteinMPNN_43636867728028
// MI455X (gfx1250) — compile-verified, matrix operand order checked by reading, not run on hardware
//
#include <hip/hip_runtime.h>

// ---------------------------------------------------------------------------
// ProteinMPNN forward on MI455X (gfx1250, wave32, WMMA).
// All dense layers -> bf16 V_WMMA_F32_16X16X32_BF16 GEMMs with fused
// bias+GELU epilogues. Weights are transpose-converted to bf16 [N][K] so both
// A and B WMMA fragments are contiguous 16B LDS reads (ds_load_b128), no
// per-element packing in the inner loop. Intermediates stay L2-resident.
// ---------------------------------------------------------------------------

typedef unsigned short u16;
typedef __attribute__((ext_vector_type(16))) __bf16 v16bf;
typedef __attribute__((ext_vector_type(8)))  float  v8f;

union BF16x16 { v16bf v; u16 u[16]; };
union F32x8   { v8f  v; float f[8]; };

__device__ inline u16 f2bf(float f) {
    unsigned u = __builtin_bit_cast(unsigned, f);
    u += 0x7FFFu + ((u >> 16) & 1u);          // round-to-nearest-even
    return (u16)(u >> 16);
}
__device__ inline float bf2f(u16 h) {
    return __builtin_bit_cast(float, (unsigned)h << 16);
}
__device__ inline float gelu_exact(float x) {
    return 0.5f * x * (1.0f + erff(x * 0.70710678118654752f));
}

// ---------------------------------------------------------------------------
// bf16 WMMA GEMM:  out[M,N] = A[M,Kd] @ W[Kd,N] (+bias) (+GELU)
// W is supplied TRANSPOSED as Wt[N][Kd].
// Block: 256 threads (8 waves). Block tile 128x64, wave tile 32x32 (2x2 WMMA),
// k-step 32. M % 128 == 0, Kd % 32 == 0 guaranteed by caller; N guarded.
// ---------------------------------------------------------------------------
#define GBM 128
#define GBN 64
#define GBK 32

__global__ __launch_bounds__(256) void gemm_bf16_kernel(
    const u16* __restrict__ A, const u16* __restrict__ Wt,
    const float* __restrict__ bias, float* __restrict__ outF,
    u16* __restrict__ outB, int M, int N, int Kd, int gelu_flag)
{
    __shared__ u16 As[GBM][GBK + 8];   // +8 keeps 16B row alignment, kills conflicts
    __shared__ u16 Bs[GBN][GBK + 8];   // B stored by-N, contiguous in k

    const int tid  = threadIdx.x;
    const int m0   = blockIdx.y * GBM;
    const int n0   = blockIdx.x * GBN;
    const int wave = tid >> 5;
    const int lane = tid & 31;
    const int wm   = (wave >> 1) * 32;   // wave row offset in block tile
    const int wn   = (wave & 1) * 32;    // wave col offset in block tile
    const int lr   = lane & 15;
    const int kb   = (lane >> 4) * 8;    // k-half select per ISA 16-bit layout

    F32x8 acc[2][2];
#pragma unroll
    for (int r = 0; r < 2; ++r)
#pragma unroll
        for (int c = 0; c < 2; ++c)
#pragma unroll
            for (int i = 0; i < 8; ++i) acc[r][c].f[i] = 0.0f;

    for (int kt = 0; kt < Kd; kt += GBK) {
        {   // A tile: 128x32 bf16, 32B per thread
            int row = tid >> 1, c0 = (tid & 1) * 16;
            const u16* src = A + (size_t)(m0 + row) * Kd + kt + c0;
            uint4 p0 = ((const uint4*)src)[0];
            uint4 p1 = ((const uint4*)src)[1];
            *((uint4*)&As[row][c0])     = p0;
            *((uint4*)&As[row][c0 + 8]) = p1;
            if (kt + GBK < Kd) __builtin_prefetch(src + GBK, 0, 1);  // next k-tile
        }
        {   // B tile: 64 N-rows x 32 k, 16B per thread (zero-fill beyond N)
            int row = tid >> 2, c0 = (tid & 3) * 8;
            int gn = n0 + row;
            uint4 p = {0u, 0u, 0u, 0u};
            if (gn < N) {
                const u16* src = Wt + (size_t)gn * Kd + kt + c0;
                p = *((const uint4*)src);
                if (kt + GBK < Kd) __builtin_prefetch(src + GBK, 0, 1);
            }
            *((uint4*)&Bs[row][c0]) = p;
        }
        __syncthreads();

        BF16x16 afr[2], bfr[2];
#pragma unroll
        for (int t = 0; t < 2; ++t) {
            int ar = wm + t * 16 + lr;         // A: lane holds row, k split by lane[4]
#pragma unroll
            for (int i = 0; i < 8; ++i) {
                afr[t].u[i]     = As[ar][kb + i];
                afr[t].u[8 + i] = As[ar][kb + 16 + i];
            }
            int bc = wn + t * 16 + lr;         // B: lane holds col, k contiguous
#pragma unroll
            for (int i = 0; i < 8; ++i) {
                bfr[t].u[i]     = Bs[bc][kb + i];
                bfr[t].u[8 + i] = Bs[bc][kb + 16 + i];
            }
        }
#pragma unroll
        for (int r = 0; r < 2; ++r)
#pragma unroll
            for (int c = 0; c < 2; ++c)
                acc[r][c].v = __builtin_amdgcn_wmma_f32_16x16x32_bf16(
                    false, afr[r].v, false, bfr[c].v,
                    (short)0, acc[r][c].v, false, false);
        __syncthreads();
    }

    // Epilogue: C/D layout lane -> (N = lane&15, M = i + 8*(lane>>4))
#pragma unroll
    for (int r = 0; r < 2; ++r)
#pragma unroll
        for (int c = 0; c < 2; ++c) {
            int col = n0 + wn + c * 16 + lr;
            if (col < N) {
                float bv = bias ? bias[col] : 0.0f;
#pragma unroll
                for (int i = 0; i < 8; ++i) {
                    int row = m0 + wm + r * 16 + kb + i;
                    float v = acc[r][c].f[i] + bv;
                    if (gelu_flag) v = gelu_exact(v);
                    size_t o = (size_t)row * N + col;
                    if (outF) outF[o] = v;
                    if (outB) outB[o] = f2bf(v);
                }
            }
        }
}

// ---------------------------------------------------------------------------
// Geometry: N,Ca,C,O -> + virtual Cb, atoms[B*L][5][3]
// ---------------------------------------------------------------------------
__global__ void atoms_kernel(const float* __restrict__ X, float* __restrict__ A, int total)
{
    int n = blockIdx.x * blockDim.x + threadIdx.x;
    if (n >= total) return;
    const float* x = X + (size_t)n * 12;
    float Nx = x[0], Ny = x[1], Nz = x[2];
    float Cax = x[3], Cay = x[4], Caz = x[5];
    float Cx = x[6], Cy = x[7], Cz = x[8];
    float Ox = x[9], Oy = x[10], Oz = x[11];
    float bx = Cax - Nx, by = Cay - Ny, bz = Caz - Nz;
    float cx = Cx - Cax, cy = Cy - Cay, cz = Cz - Caz;
    float ax = by * cz - bz * cy, ay = bz * cx - bx * cz, az = bx * cy - by * cx;
    float Cbx = -0.58273431f * ax + 0.56802827f * bx - 0.54067466f * cx + Cax;
    float Cby = -0.58273431f * ay + 0.56802827f * by - 0.54067466f * cy + Cay;
    float Cbz = -0.58273431f * az + 0.56802827f * bz - 0.54067466f * cz + Caz;
    float* o = A + (size_t)n * 15;
    o[0] = Nx;  o[1] = Ny;  o[2] = Nz;
    o[3] = Cax; o[4] = Cay; o[5] = Caz;
    o[6] = Cx;  o[7] = Cy;  o[8] = Cz;
    o[9] = Ox;  o[10] = Oy; o[11] = Oz;
    o[12] = Cbx; o[13] = Cby; o[14] = Cbz;
}

// ---------------------------------------------------------------------------
// Top-K (K=32) nearest Ca neighbors per residue. One block (256 thr) per (b,i).
// ---------------------------------------------------------------------------
__global__ __launch_bounds__(256) void topk_kernel(
    const float* __restrict__ atoms, const float* __restrict__ mask,
    int* __restrict__ E_idx, float* __restrict__ Dnb, int L)
{
    __shared__ float Drow[1024];
    __shared__ float sval[256];
    __shared__ int   sidx[256];
    int bi = blockIdx.x;               // b*L + i
    int b = bi / L;
    int t = threadIdx.x;
    const float* Ai = atoms + (size_t)bi * 15 + 3;   // Ca of i
    float xi = Ai[0], yi = Ai[1], zi = Ai[2];
    float mi = mask[bi];
    float lmax = 0.0f;
    for (int j = t; j < L; j += 256) {
        const float* Aj = atoms + (size_t)(b * L + j) * 15 + 3;
        float dx = xi - Aj[0], dy = yi - Aj[1], dz = zi - Aj[2];
        float m2 = mi * mask[b * L + j];
        float d = m2 * sqrtf(dx * dx + dy * dy + dz * dz + 1e-6f);
        Drow[j] = d;
        lmax = fmaxf(lmax, d);
    }
    sval[t] = lmax; __syncthreads();
    for (int s = 128; s > 0; s >>= 1) {
        if (t < s) sval[t] = fmaxf(sval[t], sval[t + s]);
        __syncthreads();
    }
    float dmax = sval[0]; __syncthreads();
    for (int j = t; j < L; j += 256) {
        float m2 = mi * mask[b * L + j];
        Drow[j] += (1.0f - m2) * dmax;       // D_adjust
    }
    __syncthreads();
    for (int k = 0; k < 32; ++k) {
        float bv = 3.0e38f; int bj = L;
        for (int j = t; j < L; j += 256) {
            float v = Drow[j];
            if (v < bv || (v == bv && j < bj)) { bv = v; bj = j; }
        }
        sval[t] = bv; sidx[t] = bj; __syncthreads();
        for (int s = 128; s > 0; s >>= 1) {
            if (t < s) {
                float v = sval[t + s]; int j = sidx[t + s];
                if (v < sval[t] || (v == sval[t] && j < sidx[t])) { sval[t] = v; sidx[t] = j; }
            }
            __syncthreads();
        }
        if (t == 0) {
            int j = sidx[0];
            E_idx[(size_t)bi * 32 + k] = j;
            Dnb [(size_t)bi * 32 + k] = sval[0];
            Drow[j] = 3.4e38f;
        }
        __syncthreads();
    }
}

// ---------------------------------------------------------------------------
// Edge features: 16 positional (one-hot @ pos_W == row select) + 25x16 RBF.
// Output bf16 [NE, 416] ready for the edge_W GEMM.
// ---------------------------------------------------------------------------
__constant__ int PA[24] = {0,2,3,4,1,1,1,1,0,0,0,4,4,3,0,2,3,4,2,3,4,2,3,2};
__constant__ int PB[24] = {0,2,3,4,0,2,3,4,2,3,4,2,3,2,1,1,1,1,0,0,0,4,4,3};

__global__ __launch_bounds__(256) void edge_feat_kernel(
    const float* __restrict__ atoms, const int* __restrict__ E_idx,
    const float* __restrict__ Dnb, const int* __restrict__ residue_idx,
    const int* __restrict__ chain_labels, const float* __restrict__ pos_W,
    const float* __restrict__ pos_b, u16* __restrict__ ef, int L, int total)
{
    int e = blockIdx.x * 256 + threadIdx.x;
    if (e >= total) return;
    int ni = e >> 5;
    int b = ni / L;
    int nj = b * L + E_idx[e];
    int off = residue_idx[ni] - residue_idx[nj];
    int same = (chain_labels[ni] == chain_labels[nj]);
    int dp = same ? min(max(off + 32, 0), 64) : 65;
    u16* o = ef + (size_t)e * 416;
#pragma unroll
    for (int t = 0; t < 16; ++t) o[t] = f2bf(pos_W[dp * 16 + t] + pos_b[t]);
    const float* Ai = atoms + (size_t)ni * 15;
    const float* Aj = atoms + (size_t)nj * 15;
    float ds[25];
    ds[0] = Dnb[e];
#pragma unroll
    for (int p = 0; p < 24; ++p) {
        const float* u = Ai + PA[p] * 3;
        const float* v = Aj + PB[p] * 3;
        float dx = u[0] - v[0], dy = u[1] - v[1], dz = u[2] - v[2];
        ds[p + 1] = sqrtf(dx * dx + dy * dy + dz * dz + 1e-6f);
    }
    const float inv_sig = 1.0f / 1.25f;
#pragma unroll
    for (int p = 0; p < 25; ++p) {
        float d = ds[p];
#pragma unroll
        for (int r = 0; r < 16; ++r) {
            float mu = 2.0f + (20.0f / 15.0f) * r;
            float z = (d - mu) * inv_sig;
            o[16 + p * 16 + r] = f2bf(__expf(-z * z));
        }
    }
}

// ---------------------------------------------------------------------------
// LayerNorm over H=128: y = LN(res + alpha*x)*g + b, optional row mask.
// One 128-thread block per row; writes fp32 and/or bf16.
// ---------------------------------------------------------------------------
__global__ __launch_bounds__(128) void ln_kernel(
    const float* __restrict__ x, const float* __restrict__ res, float alpha,
    const float* __restrict__ g, const float* __restrict__ bvec,
    const float* __restrict__ rowmask, float* __restrict__ outF,
    u16* __restrict__ outB)
{
    __shared__ float s1[128], s2[128];
    int r = blockIdx.x, t = threadIdx.x;
    size_t o = (size_t)r * 128 + t;
    float v = alpha * x[o] + (res ? res[o] : 0.0f);
    s1[t] = v; s2[t] = v * v; __syncthreads();
    for (int s = 64; s > 0; s >>= 1) {
        if (t < s) { s1[t] += s1[t + s]; s2[t] += s2[t + s]; }
        __syncthreads();
    }
    float mu = s1[0] * (1.0f / 128.0f);
    float var = s2[0] * (1.0f / 128.0f) - mu * mu;
    float y = (v - mu) * rsqrtf(var + 1e-5f) * g[t] + bvec[t];
    if (rowmask) y *= rowmask[r];
    if (outF) outF[o] = y;
    if (outB) outB[o] = f2bf(y);
}

// Sum messages over K neighbors (optionally * mask_attend).
__global__ __launch_bounds__(128) void summsg_kernel(
    const float* __restrict__ msg, const float* __restrict__ att,
    float* __restrict__ out)
{
    int r = blockIdx.x, t = threadIdx.x;
    float s = 0.0f;
    for (int k = 0; k < 32; ++k) {
        float m = att ? att[r * 32 + k] : 1.0f;
        s += m * msg[((size_t)r * 32 + k) * 128 + t];
    }
    out[(size_t)r * 128 + t] = s;
}

// h_EV = [h_V[i], h_E[e], h_V[j]] in bf16 (encoder, 384 dims).
__global__ __launch_bounds__(128) void concat_enc_kernel(
    const float* __restrict__ hV, const float* __restrict__ hE,
    const int* __restrict__ E_idx, u16* __restrict__ hEV, int L)
{
    int e = blockIdx.x, t = threadIdx.x;
    int ni = e >> 5, b = ni / L;
    int nj = b * L + E_idx[e];
    u16* o = hEV + (size_t)e * 384;
    o[t]       = f2bf(hV[(size_t)ni * 128 + t]);
    o[128 + t] = f2bf(hE[(size_t)e  * 128 + t]);
    o[256 + t] = f2bf(hV[(size_t)nj * 128 + t]);
}

__global__ void attmask_kernel(const float* __restrict__ mask,
                               const int* __restrict__ E_idx,
                               float* __restrict__ att, int L, int total)
{
    int e = blockIdx.x * blockDim.x + threadIdx.x;
    if (e >= total) return;
    int ni = e >> 5, b = ni / L;
    att[e] = mask[ni] * mask[b * L + E_idx[e]];
}

// Autoregressive rank = position in stable argsort of keys (O(L) per thread).
__global__ void rank_kernel(const float* __restrict__ mask,
                            const float* __restrict__ chain_M,
                            const float* __restrict__ randn,
                            int* __restrict__ rankb, int L, int total)
{
    int n = blockIdx.x * blockDim.x + threadIdx.x;
    if (n >= total) return;
    int b = n / L, i = n % L;
    float ki = (chain_M[n] * mask[n] + 1e-4f) * fabsf(randn[n]);
    int r = 0;
    for (int j = 0; j < L; ++j) {
        int nj = b * L + j;
        float kj = (chain_M[nj] * mask[nj] + 1e-4f) * fabsf(randn[nj]);
        r += (kj < ki) || (kj == ki && j < i);
    }
    rankb[n] = r;
}

__global__ __launch_bounds__(128) void hs_kernel(
    const int* __restrict__ S, const float* __restrict__ Ws, float* __restrict__ hS)
{
    int n = blockIdx.x, t = threadIdx.x;
    hS[(size_t)n * 128 + t] = Ws[(size_t)S[n] * 128 + t];
}

// h_ES = [h_E, h_S[j]] bf16 (256 dims).
__global__ __launch_bounds__(128) void hes_kernel(
    const float* __restrict__ hE, const float* __restrict__ hS,
    const int* __restrict__ E_idx, u16* __restrict__ hES, int L)
{
    int e = blockIdx.x, t = threadIdx.x;
    int ni = e >> 5, b = ni / L;
    int nj = b * L + E_idx[e];
    hES[(size_t)e * 256 + t]       = f2bf(hE[(size_t)e  * 128 + t]);
    hES[(size_t)e * 256 + 128 + t] = f2bf(hS[(size_t)nj * 128 + t]);
}

// h_EXV_fw = mask_fw * [h_E, 0, h_V_enc[j]] bf16 (384), plus mask_bw.
__global__ __launch_bounds__(128) void buildfw_kernel(
    const float* __restrict__ hE, const float* __restrict__ hV,
    const int* __restrict__ E_idx, const int* __restrict__ rankb,
    const float* __restrict__ mask, u16* __restrict__ hEXV,
    float* __restrict__ mbw, int L)
{
    int e = blockIdx.x, t = threadIdx.x;
    int ni = e >> 5, b = ni / L;
    int nj = b * L + E_idx[e];
    float att = (rankb[ni] > rankb[nj]) ? 1.0f : 0.0f;
    float mi = mask[ni];
    float fw = mi * (1.0f - att);
    if (t == 0) mbw[e] = mi * att;
    u16* o = hEXV + (size_t)e * 384;
    o[t]       = f2bf(fw * hE[(size_t)e * 128 + t]);
    o[128 + t] = 0;
    o[256 + t] = f2bf(fw * hV[(size_t)nj * 128 + t]);
}

// Decoder h_EV = [h_V[i], mask_bw*[h_ES, h_V[j]] + h_EXV_fw] bf16 (512).
__global__ __launch_bounds__(128) void concat_dec_kernel(
    const float* __restrict__ hV, const u16* __restrict__ hES,
    const u16* __restrict__ hEXV, const float* __restrict__ mbw,
    const int* __restrict__ E_idx, u16* __restrict__ hEV, int L)
{
    int e = blockIdx.x, t = threadIdx.x;
    int ni = e >> 5, b = ni / L;
    int nj = b * L + E_idx[e];
    u16* o = hEV + (size_t)e * 512;
    o[t] = f2bf(hV[(size_t)ni * 128 + t]);
    float bw = mbw[e];
#pragma unroll
    for (int c = 0; c < 3; ++c) {
        int d = c * 128 + t;
        float esv = (d < 256) ? bf2f(hES[(size_t)e * 256 + d])
                              : hV[(size_t)nj * 128 + (d - 256)];
        float v = bw * esv + bf2f(hEXV[(size_t)e * 384 + d]);
        o[128 + d] = f2bf(v);
    }
}

__global__ void logsoftmax_kernel(const float* __restrict__ logits,
                                  const float* __restrict__ bias,
                                  float* __restrict__ out, int total)
{
    int n = blockIdx.x * blockDim.x + threadIdx.x;
    if (n >= total) return;
    float v[21]; float mx = -3.0e38f;
#pragma unroll
    for (int c = 0; c < 21; ++c) { v[c] = logits[(size_t)n * 32 + c] + bias[c]; mx = fmaxf(mx, v[c]); }
    float s = 0.0f;
#pragma unroll
    for (int c = 0; c < 21; ++c) s += __expf(v[c] - mx);
    float ls = logf(s) + mx;
#pragma unroll
    for (int c = 0; c < 21; ++c) out[(size_t)n * 21 + c] = v[c] - ls;
}

// Transpose-convert weights: src is fp32 [K][N] -> dst bf16 [N][K].
__global__ void cvtT_kernel(const float* __restrict__ src, u16* __restrict__ dst,
                            int K, int N)
{
    int i = blockIdx.x * blockDim.x + threadIdx.x;
    if (i >= K * N) return;
    int k = i / N, n = i % N;
    dst[(size_t)n * K + k] = f2bf(src[i]);
}

// Wout (128x21) -> transposed + padded bf16 [32][128], zero rows 21..31.
__global__ void padwT_kernel(const float* __restrict__ src, u16* __restrict__ dst)
{
    int i = blockIdx.x * blockDim.x + threadIdx.x;
    if (i >= 32 * 128) return;
    int c = i >> 7, r = i & 127;           // c = output class, r = hidden dim
    dst[i] = (c < 21) ? f2bf(src[r * 21 + c]) : (u16)0;
}

__global__ void fill0_kernel(float* __restrict__ p, int n)
{
    int i = blockIdx.x * blockDim.x + threadIdx.x;
    if (i < n) p[i] = 0.0f;
}

// ---------------------------------------------------------------------------
// Host orchestration
// ---------------------------------------------------------------------------
extern "C" void kernel_launch(void* const* d_in, const int* in_sizes, int n_in,
                              void* d_out, int out_size, void* d_ws, size_t ws_size,
                              hipStream_t stream)
{
    (void)n_in; (void)out_size; (void)ws_size;
    const int L = 1024, H = 128;
    const int BL = in_sizes[2];       // B*L
    const int NE = BL * 32;           // B*L*K edge tokens

    const float* X            = (const float*)d_in[0];
    const int*   S            = (const int*)  d_in[1];
    const float* mask         = (const float*)d_in[2];
    const float* chain_M      = (const float*)d_in[3];
    const int*   residue_idx  = (const int*)  d_in[4];
    const int*   chain_labels = (const int*)  d_in[5];
    const float* randn        = (const float*)d_in[6];

    // params flattened in jax pytree order (alphabetical dict keys)
    int pi = 7;
    auto F = [&](int i) { return (const float*)d_in[i]; };
    const float* We_W   = F(pi++);
    const float* We_b   = F(pi++);
    const float* Wout_W = F(pi++);
    const float* Wout_b = F(pi++);
    const float* Ws     = F(pi++);
    struct DecP { const float *W1,*W2,*W3,*b1,*b2,*b3,*ffWi,*ffWo,*ffbi,*ffbo,*n1b,*n1g,*n2b,*n2g; } dp_[3];
    for (int l = 0; l < 3; ++l) {
        dp_[l].W1 = F(pi++); dp_[l].W2 = F(pi++); dp_[l].W3 = F(pi++);
        dp_[l].b1 = F(pi++); dp_[l].b2 = F(pi++); dp_[l].b3 = F(pi++);
        dp_[l].ffWi = F(pi++); dp_[l].ffWo = F(pi++); dp_[l].ffbi = F(pi++); dp_[l].ffbo = F(pi++);
        dp_[l].n1b = F(pi++); dp_[l].n1g = F(pi++); dp_[l].n2b = F(pi++); dp_[l].n2g = F(pi++);
    }
    struct EncP { const float *W1,*W2,*W3,*b1,*b2,*b3,*eW1,*eW2,*eW3,*eb1,*eb2,*eb3,
                              *ffWi,*ffWo,*ffbi,*ffbo,*n1b,*n1g,*n2b,*n2g,*n3b,*n3g; } ep[3];
    for (int l = 0; l < 3; ++l) {
        ep[l].W1 = F(pi++);  ep[l].W2 = F(pi++);  ep[l].W3 = F(pi++);
        ep[l].b1 = F(pi++);  ep[l].b2 = F(pi++);  ep[l].b3 = F(pi++);
        ep[l].eW1 = F(pi++); ep[l].eW2 = F(pi++); ep[l].eW3 = F(pi++);
        ep[l].eb1 = F(pi++); ep[l].eb2 = F(pi++); ep[l].eb3 = F(pi++);
        ep[l].ffWi = F(pi++); ep[l].ffWo = F(pi++); ep[l].ffbi = F(pi++); ep[l].ffbo = F(pi++);
        ep[l].n1b = F(pi++); ep[l].n1g = F(pi++); ep[l].n2b = F(pi++); ep[l].n2g = F(pi++);
        ep[l].n3b = F(pi++); ep[l].n3g = F(pi++);
    }
    const float* feat_edge_W = F(pi++);
    const float* feat_norm_b = F(pi++);
    const float* feat_norm_g = F(pi++);
    const float* feat_pos_W  = F(pi++);
    const float* feat_pos_b  = F(pi++);

    // ---- workspace bump allocator (256B aligned) ----
    char* base = (char*)d_ws;
    size_t off = 0;
    auto alloc = [&](size_t bytes) -> void* {
        size_t o = (off + 255) & ~(size_t)255;
        off = o + bytes;
        return (void*)(base + o);
    };

    float* atoms  = (float*)alloc((size_t)BL * 15 * 4);
    int*   E_idx  = (int*)  alloc((size_t)NE * 4);
    float* Dnb    = (float*)alloc((size_t)NE * 4);
    float* attm   = (float*)alloc((size_t)NE * 4);
    float* mbw    = (float*)alloc((size_t)NE * 4);
    int*   rankb  = (int*)  alloc((size_t)BL * 4);
    u16*   ef     = (u16*)  alloc((size_t)NE * 416 * 2);
    float* hE     = (float*)alloc((size_t)NE * H * 4);
    float* msg    = (float*)alloc((size_t)NE * H * 4);
    u16*   mlp1   = (u16*)  alloc((size_t)NE * H * 2);
    u16*   mlp2   = (u16*)  alloc((size_t)NE * H * 2);
    u16*   hEV    = (u16*)  alloc((size_t)NE * 512 * 2);
    u16*   hES    = (u16*)  alloc((size_t)NE * 256 * 2);
    u16*   hEXV   = (u16*)  alloc((size_t)NE * 384 * 2);
    float* hV     = (float*)alloc((size_t)BL * H * 4);
    u16*   hVb    = (u16*)  alloc((size_t)BL * H * 2);
    float* hV2    = (float*)alloc((size_t)BL * H * 4);
    u16*   hV2b   = (u16*)  alloc((size_t)BL * H * 2);
    float* vsum   = (float*)alloc((size_t)BL * H * 4);
    float* hS     = (float*)alloc((size_t)BL * H * 4);
    u16*   ffnmid = (u16*)  alloc((size_t)BL * 512 * 2);
    float* ffnout = (float*)alloc((size_t)BL * H * 4);
    float* logits = (float*)alloc((size_t)BL * 32 * 4);

    // ---- bf16 transpose-convert weights (deterministic, every call) ----
    auto cvtW = [&](const float* src, int K, int N) -> u16* {
        u16* d = (u16*)alloc((size_t)K * N * 2);
        cvtT_kernel<<<(K * N + 255) / 256, 256, 0, stream>>>(src, d, K, N);
        return d;
    };
    u16* edgeWb = cvtW(feat_edge_W, 416, 128);
    u16* WeWb   = cvtW(We_W, 128, 128);
    struct EncB { u16 *W1,*W2,*W3,*eW1,*eW2,*eW3,*ffWi,*ffWo; } eb[3];
    for (int l = 0; l < 3; ++l) {
        eb[l].W1  = cvtW(ep[l].W1, 384, 128);
        eb[l].W2  = cvtW(ep[l].W2, 128, 128);
        eb[l].W3  = cvtW(ep[l].W3, 128, 128);
        eb[l].eW1 = cvtW(ep[l].eW1, 384, 128);
        eb[l].eW2 = cvtW(ep[l].eW2, 128, 128);
        eb[l].eW3 = cvtW(ep[l].eW3, 128, 128);
        eb[l].ffWi = cvtW(ep[l].ffWi, 128, 512);
        eb[l].ffWo = cvtW(ep[l].ffWo, 512, 128);
    }
    struct DecB { u16 *W1,*W2,*W3,*ffWi,*ffWo; } db[3];
    for (int l = 0; l < 3; ++l) {
        db[l].W1 = cvtW(dp_[l].W1, 512, 128);
        db[l].W2 = cvtW(dp_[l].W2, 128, 128);
        db[l].W3 = cvtW(dp_[l].W3, 128, 128);
        db[l].ffWi = cvtW(dp_[l].ffWi, 128, 512);
        db[l].ffWo = cvtW(dp_[l].ffWo, 512, 128);
    }
    u16* WoutPad = (u16*)alloc(32 * 128 * 2);
    padwT_kernel<<<(32 * 128 + 255) / 256, 256, 0, stream>>>(Wout_W, WoutPad);

    auto gemm = [&](const u16* A, const u16* Wt, const float* bias, float* oF,
                    u16* oB, int M, int N, int Kd, int gelu) {
        dim3 g((N + GBN - 1) / GBN, M / GBM);
        gemm_bf16_kernel<<<g, 256, 0, stream>>>(A, Wt, bias, oF, oB, M, N, Kd, gelu);
    };
    auto ln = [&](const float* x, const float* res, float alpha, const float* g_,
                  const float* b_, const float* rm, float* oF, u16* oB, int M) {
        ln_kernel<<<M, 128, 0, stream>>>(x, res, alpha, g_, b_, rm, oF, oB);
    };

    // ---- features ----
    atoms_kernel<<<(BL + 255) / 256, 256, 0, stream>>>(X, atoms, BL);
    topk_kernel<<<BL, 256, 0, stream>>>(atoms, mask, E_idx, Dnb, L);
    edge_feat_kernel<<<(NE + 255) / 256, 256, 0, stream>>>(
        atoms, E_idx, Dnb, residue_idx, chain_labels, feat_pos_W, feat_pos_b, ef, L, NE);
    gemm(ef, edgeWb, nullptr, msg, nullptr, NE, H, 416, 0);       // E = edge_feat @ edge_W
    ln(msg, nullptr, 1.0f, feat_norm_g, feat_norm_b, nullptr, nullptr, mlp1, NE);
    gemm(mlp1, WeWb, We_b, hE, nullptr, NE, H, H, 0);             // h_E
    fill0_kernel<<<(BL * H + 255) / 256, 256, 0, stream>>>(hV, BL * H);
    attmask_kernel<<<(NE + 255) / 256, 256, 0, stream>>>(mask, E_idx, attm, L, NE);

    // ---- encoder ----
    for (int l = 0; l < 3; ++l) {
        concat_enc_kernel<<<NE, 128, 0, stream>>>(hV, hE, E_idx, hEV, L);
        gemm(hEV,  eb[l].W1, ep[l].b1, nullptr, mlp1, NE, H, 384, 1);
        gemm(mlp1, eb[l].W2, ep[l].b2, nullptr, mlp2, NE, H, H, 1);
        gemm(mlp2, eb[l].W3, ep[l].b3, msg, nullptr, NE, H, H, 0);
        summsg_kernel<<<BL, 128, 0, stream>>>(msg, attm, vsum);
        ln(vsum, hV, 1.0f / 30.0f, ep[l].n1g, ep[l].n1b, nullptr, hV2, hV2b, BL);
        gemm(hV2b,   eb[l].ffWi, ep[l].ffbi, nullptr, ffnmid, BL, 512, H, 1);
        gemm(ffnmid, eb[l].ffWo, ep[l].ffbo, ffnout, nullptr, BL, H, 512, 0);
        ln(ffnout, hV2, 1.0f, ep[l].n2g, ep[l].n2b, mask, hV, hVb, BL);
        // edge update
        concat_enc_kernel<<<NE, 128, 0, stream>>>(hV, hE, E_idx, hEV, L);
        gemm(hEV,  eb[l].eW1, ep[l].eb1, nullptr, mlp1, NE, H, 384, 1);
        gemm(mlp1, eb[l].eW2, ep[l].eb2, nullptr, mlp2, NE, H, H, 1);
        gemm(mlp2, eb[l].eW3, ep[l].eb3, msg, nullptr, NE, H, H, 0);
        ln(msg, hE, 1.0f, ep[l].n3g, ep[l].n3b, nullptr, hE, nullptr, NE);
    }

    // ---- decoder prep ----
    hs_kernel<<<BL, 128, 0, stream>>>(S, Ws, hS);
    rank_kernel<<<(BL + 255) / 256, 256, 0, stream>>>(mask, chain_M, randn, rankb, L, BL);
    hes_kernel<<<NE, 128, 0, stream>>>(hE, hS, E_idx, hES, L);
    buildfw_kernel<<<NE, 128, 0, stream>>>(hE, hV, E_idx, rankb, mask, hEXV, mbw, L);

    // ---- decoder ----
    for (int l = 0; l < 3; ++l) {
        concat_dec_kernel<<<NE, 128, 0, stream>>>(hV, hES, hEXV, mbw, E_idx, hEV, L);
        gemm(hEV,  db[l].W1, dp_[l].b1, nullptr, mlp1, NE, H, 512, 1);
        gemm(mlp1, db[l].W2, dp_[l].b2, nullptr, mlp2, NE, H, H, 1);
        gemm(mlp2, db[l].W3, dp_[l].b3, msg, nullptr, NE, H, H, 0);
        summsg_kernel<<<BL, 128, 0, stream>>>(msg, nullptr, vsum);
        ln(vsum, hV, 1.0f / 30.0f, dp_[l].n1g, dp_[l].n1b, nullptr, hV2, hV2b, BL);
        gemm(hV2b,   db[l].ffWi, dp_[l].ffbi, nullptr, ffnmid, BL, 512, H, 1);
        gemm(ffnmid, db[l].ffWo, dp_[l].ffbo, ffnout, nullptr, BL, H, 512, 0);
        ln(ffnout, hV2, 1.0f, dp_[l].n2g, dp_[l].n2b, mask, hV, hVb, BL);
    }

    // ---- output head ----
    gemm(hVb, WoutPad, nullptr, logits, nullptr, BL, 32, H, 0);
    logsoftmax_kernel<<<(BL + 255) / 256, 256, 0, stream>>>(logits, Wout_b, (float*)d_out, BL);
}
